// tru_53214644798208
// MI455X (gfx1250) — matrix operand order repaired and checked by reading, not compiled, not run on hardware
//
#include <hip/hip_runtime.h>
#include <stdint.h>

typedef __attribute__((ext_vector_type(16))) __bf16 v16bf;
typedef __attribute__((ext_vector_type(8)))  __bf16 v8bf;
typedef __attribute__((ext_vector_type(8)))  float  v8f;
typedef __attribute__((ext_vector_type(4)))  float  f32x4;

static constexpr int B_   = 32;
static constexpr int NHID = 1024;
static constexpr int L_   = 2048;
static constexpr int NHEAD= 16;

// ---------------------------------------------------------------------------
// fp32 -> bf16 weight conversion (WK_w / WV_w, 1M elems each, run once)
// ---------------------------------------------------------------------------
__global__ __launch_bounds__(256) void k_cvt(const float* __restrict__ in,
                                             __bf16* __restrict__ out) {
  int idx = (blockIdx.x * 256 + threadIdx.x) * 4;
  f32x4 v = *(const f32x4*)(in + idx);
  out[idx + 0] = (__bf16)v[0];
  out[idx + 1] = (__bf16)v[1];
  out[idx + 2] = (__bf16)v[2];
  out[idx + 3] = (__bf16)v[3];
}

// ---------------------------------------------------------------------------
// q = x @ WQ^T + bq       (32x1024, trivial)
// ---------------------------------------------------------------------------
__global__ __launch_bounds__(256) void k_qproj(const float* __restrict__ x,
                                               const float* __restrict__ Wq,
                                               const float* __restrict__ bq,
                                               float* __restrict__ q) {
  int idx = blockIdx.x * 256 + threadIdx.x;            // 32768 outputs
  int b = idx >> 10, o = idx & 1023;
  const float* xr = x + b * NHID;
  const float* wr = Wq + (size_t)o * NHID;
  float s = bq[o];
  for (int c = 0; c < NHID; c += 4) {
    f32x4 xv = *(const f32x4*)(xr + c);
    f32x4 wv = *(const f32x4*)(wr + c);
    s += xv[0]*wv[0] + xv[1]*wv[1] + xv[2]*wv[2] + xv[3]*wv[3];
  }
  q[idx] = s;
}

// ---------------------------------------------------------------------------
// Projection GEMM (K or V):  O[b][o][l] = bias[o] + sum_c W[o][c] * Y[b][c][l]
// M=1024, N=2048, K=1024 per batch.  bf16 WMMA, fp32 accumulate.
// WG: 256 thr (8 waves), tile 128x128, wave = 64x32 (4x2 wmma accums).
// A-tile (bf16 weights): GLOBAL_LOAD_ASYNC_TO_LDS_B128, double buffered.
// B-tile (y fp32): register stage + cvt + transpose into n-major LDS.
// ---------------------------------------------------------------------------
__global__ __launch_bounds__(256) void k_proj(const __bf16* __restrict__ Wb,
                                              const float* __restrict__ bias,
                                              const float* __restrict__ Y,
                                              __bf16* __restrict__ O) {
  constexpr int KT = 32, NSTEP = NHID / KT;
  constexpr int AS = 40;            // padded LDS row strides (bf16 elems)
  constexpr int BS = 40;
  __shared__ __attribute__((aligned(16))) __bf16 As[2][128 * AS]; // [m][k]
  __shared__ __attribute__((aligned(16))) __bf16 Bs[128 * BS];    // [n][k]

  const int tid  = threadIdx.x;
  const int b    = blockIdx.z;
  const int m0   = blockIdx.y * 128;
  const int n0   = blockIdx.x * 128;
  const int lane = tid & 31, wid = tid >> 5;
  const int wm   = wid & 1,  wn  = wid >> 1;
  const int r16  = lane & 15, khalf = lane >> 4;

  const float* Yb = Y + (size_t)b * NHID * L_;
  __bf16*      Ob = O + (size_t)b * NHID * L_;

  // A-tile: 128 rows x 64B (KT=32 bf16) = 512 x 16B chunks, async DMA to LDS
  auto acopy = [&](int kk, int buf) {
    for (int i = 0; i < 2; ++i) {
      int slot = tid + i * 256;               // 0..511
      int row = slot >> 2, c16 = slot & 3;
      const __bf16* gp = Wb + (size_t)(m0 + row) * NHID + kk + c16 * 8;
      unsigned la = (unsigned)(uintptr_t)&As[buf][row * AS + c16 * 8];
      asm volatile("global_load_async_to_lds_b128 %0, %1, off"
                   :: "v"(la), "v"(gp) : "memory");
    }
  };

  f32x4 rb[4];                                // B staging regs (held across wmma)
  auto gloadB = [&](int kk) {
    for (int i = 0; i < 4; ++i) {             // B tile: 32x128 fp32 (row k, contig l)
      int slot = tid + i * 256;
      int k = slot >> 5, n4 = slot & 31;
      rb[i] = *(const f32x4*)(Yb + (size_t)(kk + k) * L_ + n0 + n4 * 4);
    }
  };
  auto lstoreB = [&]() {
    for (int i = 0; i < 4; ++i) {
      int slot = tid + i * 256;
      int k = slot >> 5, n4 = slot & 31;
      for (int j = 0; j < 4; ++j)             // transpose into n-major
        Bs[(n4 * 4 + j) * BS + k] = (__bf16)rb[i][j];
    }
  };

  v8f acc[4][2];
  for (int sm = 0; sm < 4; ++sm)
    for (int sn = 0; sn < 2; ++sn)
      for (int e = 0; e < 8; ++e) acc[sm][sn][e] = 0.f;

  acopy(0, 0);
  gloadB(0);
  for (int s = 0; s < NSTEP; ++s) {
    __syncthreads();                          // compute(s-1) done with Bs & As[(s+1)&1]
    lstoreB();
    if (s + 1 < NSTEP) {
      acopy((s + 1) * KT, (s + 1) & 1);       // DMA next A tile (overlaps wmma)
      asm volatile("s_wait_asynccnt 2" ::: "memory"); // in-order: current 2 landed
    } else {
      asm volatile("s_wait_asynccnt 0" ::: "memory");
    }
    __syncthreads();                          // Bs stores + As[s&1] visible to all waves
    if (s + 1 < NSTEP) gloadB((s + 1) * KT);  // global B loads in flight during wmma
    if (s + 2 < NSTEP)
      __builtin_prefetch(Yb + (size_t)((s + 2) * KT) * L_ + n0 + (tid & 31) * 4, 0, 1);

    const __bf16* Ab = As[s & 1];
    v16bf af[4], bfr[2];
    for (int sm = 0; sm < 4; ++sm) {          // A-frag: row=lane%16, K chunks per ISA
      int row = wm * 64 + sm * 16 + r16;
      v8bf lo = *(const v8bf*)&Ab[row * AS + khalf * 8];        // K = kh*8 .. +7
      v8bf hi = *(const v8bf*)&Ab[row * AS + 16 + khalf * 8];   // K = 16+kh*8 ..
      for (int e = 0; e < 8; ++e) { af[sm][e] = lo[e]; af[sm][e + 8] = hi[e]; }
    }
    for (int sn = 0; sn < 2; ++sn) {          // B-frag: col=lane%16, K chunks per ISA
      int col = wn * 32 + sn * 16 + r16;
      v8bf lo = *(const v8bf*)&Bs[col * BS + khalf * 8];
      v8bf hi = *(const v8bf*)&Bs[col * BS + 16 + khalf * 8];
      for (int e = 0; e < 8; ++e) { bfr[sn][e] = lo[e]; bfr[sn][e + 8] = hi[e]; }
    }
    for (int sm = 0; sm < 4; ++sm)
      for (int sn = 0; sn < 2; ++sn)
        acc[sm][sn] = __builtin_amdgcn_wmma_f32_16x16x32_bf16(
            false, af[sm], false, bfr[sn], (short)0, acc[sm][sn], false, false);
  }

  for (int sm = 0; sm < 4; ++sm) {            // epilogue: +bias, cvt bf16, store
    int mbase = m0 + wm * 64 + sm * 16 + khalf * 8;
    for (int sn = 0; sn < 2; ++sn) {
      int n = n0 + wn * 32 + sn * 16 + r16;
      for (int vg = 0; vg < 8; ++vg) {
        int m = mbase + vg;
        Ob[(size_t)m * L_ + n] = (__bf16)(acc[sm][sn][vg] + bias[m]);
      }
    }
  }
}

// ---------------------------------------------------------------------------
// Attention per (b,h): scores -> softmax -> weighted V sum; z = x + att
// ---------------------------------------------------------------------------
__global__ __launch_bounds__(256) void k_attn(const float* __restrict__ q,
                                              const __bf16* __restrict__ K,
                                              const __bf16* __restrict__ V,
                                              const float* __restrict__ x,
                                              float* __restrict__ z) {
  __shared__ float qs[64];
  __shared__ float as_[2048];
  __shared__ float red[256];
  const int tid = threadIdx.x;
  const int h = blockIdx.x, b = blockIdx.y;
  if (tid < 64) qs[tid] = q[b * NHID + h * 64 + tid];
  __syncthreads();

  const __bf16* Kb = K + (size_t)b * NHID * L_ + (size_t)h * 64 * L_;
  float s[8];
  for (int i = 0; i < 8; ++i) s[i] = 0.f;
  for (int d = 0; d < 64; ++d) {
    float qd = qs[d];
    const __bf16* kr = Kb + (size_t)d * L_;
    for (int i = 0; i < 8; ++i) s[i] += qd * (float)kr[tid + i * 256];
  }
  for (int i = 0; i < 8; ++i) s[i] *= 0.125f;   // 1/sqrt(DK)

  float m = s[0];
  for (int i = 1; i < 8; ++i) m = fmaxf(m, s[i]);
  red[tid] = m; __syncthreads();
  for (int off = 128; off > 0; off >>= 1) {
    if (tid < off) red[tid] = fmaxf(red[tid], red[tid + off]);
    __syncthreads();
  }
  m = red[0]; __syncthreads();

  float e[8], ssum = 0.f;
  for (int i = 0; i < 8; ++i) { e[i] = __expf(s[i] - m); ssum += e[i]; }
  red[tid] = ssum; __syncthreads();
  for (int off = 128; off > 0; off >>= 1) {
    if (tid < off) red[tid] += red[tid + off];
    __syncthreads();
  }
  float inv = 1.f / red[0]; __syncthreads();
  for (int i = 0; i < 8; ++i) as_[tid + i * 256] = e[i] * inv;
  __syncthreads();

  // att[d] = sum_l a[l]*V[d][l]; thread = (part 0..3) x (d 0..63)
  const int d = tid & 63, part = tid >> 6;
  const __bf16* Vr = V + (size_t)b * NHID * L_ + (size_t)(h * 64 + d) * L_ + part * 512;
  const float* ar = &as_[part * 512];
  float acc = 0.f;
  for (int j = 0; j < 512; j += 8) {
    v8bf vv = *(const v8bf*)(Vr + j);
    for (int e2 = 0; e2 < 8; ++e2) acc += ar[j + e2] * (float)vv[e2];
  }
  red[tid] = acc; __syncthreads();
  if (tid < 64) {
    float att = red[tid] + red[tid + 64] + red[tid + 128] + red[tid + 192];
    z[b * NHID + h * 64 + tid] = x[b * NHID + h * 64 + tid] + att;
  }
}

// ---------------------------------------------------------------------------
// Batchnorm (train) over batch axis, with optional residual add + per-ch bias
// ---------------------------------------------------------------------------
__global__ __launch_bounds__(256) void k_bn(const float* __restrict__ in,
                                            const float* __restrict__ addA,
                                            const float* __restrict__ cbias,
                                            const float* __restrict__ g,
                                            const float* __restrict__ bb,
                                            float* __restrict__ out) {
  int c = blockIdx.x * 256 + threadIdx.x;
  if (c >= NHID) return;
  float cb = cbias ? cbias[c] : 0.f;
  float v[32], mean = 0.f;
  for (int b = 0; b < 32; ++b) {
    float t = in[b * NHID + c] + cb;
    if (addA) t += addA[b * NHID + c];
    v[b] = t; mean += t;
  }
  mean *= (1.f / 32.f);
  float var = 0.f;
  for (int b = 0; b < 32; ++b) { float d = v[b] - mean; var += d * d; }
  var *= (1.f / 32.f);
  float rs = rsqrtf(var + 1e-5f);
  float gc = g[c], bc = bb[c];
  for (int b = 0; b < 32; ++b)
    out[b * NHID + c] = (v[b] - mean) * rs * gc + bc;
}

// ---------------------------------------------------------------------------
// MLP GEMM: Out[32][N] = act(A[32][K] @ Wt[N][K]^T + bias)
// One wave per 16x16 tile; fragments built from global fp32 (cvt to bf16).
// ---------------------------------------------------------------------------
template <int KDIM, int NDIM, bool RELU>
__global__ __launch_bounds__(256) void k_mlp(const float* __restrict__ A,
                                             const float* __restrict__ Wt,
                                             const float* __restrict__ bias,
                                             float* __restrict__ Out) {
  const int tid = threadIdx.x, lane = tid & 31, wid = tid >> 5;
  const int tile = blockIdx.x * 8 + wid;
  const int nt = tile >> 1, mt = tile & 1;
  const int r16 = lane & 15, khalf = lane >> 4;

  v8f acc;
  for (int e = 0; e < 8; ++e) acc[e] = 0.f;

  const float* arow = A  + (size_t)(mt * 16 + r16) * KDIM;
  const float* brow = Wt + (size_t)(nt * 16 + r16) * KDIM;
  for (int kk = 0; kk < KDIM; kk += 32) {
    v16bf af, bfr;
    const float* ap = arow + kk + khalf * 8;
    for (int e = 0; e < 8; ++e) { af[e] = (__bf16)ap[e]; af[e + 8] = (__bf16)ap[e + 16]; }
    const float* bp = brow + kk + khalf * 8;
    for (int e = 0; e < 8; ++e) { bfr[e] = (__bf16)bp[e]; bfr[e + 8] = (__bf16)bp[e + 16]; }
    acc = __builtin_amdgcn_wmma_f32_16x16x32_bf16(
        false, af, false, bfr, (short)0, acc, false, false);
  }
  int n = nt * 16 + r16;
  float bv = bias ? bias[n] : 0.f;
  int mbase = mt * 16 + khalf * 8;
  for (int vg = 0; vg < 8; ++vg) {
    float val = acc[vg] + bv;
    if (RELU) val = fmaxf(val, 0.f);
    Out[(size_t)(mbase + vg) * NDIM + n] = val;
  }
}

// ---------------------------------------------------------------------------
extern "C" void kernel_launch(void* const* d_in, const int* in_sizes, int n_in,
                              void* d_out, int out_size, void* d_ws, size_t ws_size,
                              hipStream_t stream) {
  const float* x    = (const float*)d_in[0];
  const float* y    = (const float*)d_in[1];
  const float* WQw  = (const float*)d_in[2];
  const float* WQb  = (const float*)d_in[3];
  const float* WKw  = (const float*)d_in[4];
  const float* WKb  = (const float*)d_in[5];
  const float* WVw  = (const float*)d_in[6];
  const float* WVb  = (const float*)d_in[7];
  const float* bn1g = (const float*)d_in[8];
  const float* bn1b = (const float*)d_in[9];
  const float* w1w  = (const float*)d_in[10];
  const float* w1b  = (const float*)d_in[11];
  const float* w2w  = (const float*)d_in[12];
  const float* w2b  = (const float*)d_in[13];
  const float* bn2g = (const float*)d_in[14];
  const float* bn2b = (const float*)d_in[15];

  char* ws = (char*)d_ws;
  size_t off = 0;
  auto alloc = [&](size_t bytes) -> void* {
    void* p = ws + off;
    off = (off + bytes + 255) & ~(size_t)255;
    return p;
  };
  __bf16* Kb   = (__bf16*)alloc((size_t)B_ * NHID * L_ * 2);  // 134 MB
  __bf16* Vb   = (__bf16*)alloc((size_t)B_ * NHID * L_ * 2);  // 134 MB
  __bf16* WKbf = (__bf16*)alloc((size_t)NHID * NHID * 2);     // 2 MB
  __bf16* WVbf = (__bf16*)alloc((size_t)NHID * NHID * 2);     // 2 MB
  float*  q    = (float*)alloc((size_t)B_ * NHID * 4);
  float*  z    = (float*)alloc((size_t)B_ * NHID * 4);
  float*  ret  = (float*)alloc((size_t)B_ * NHID * 4);
  float*  h    = (float*)alloc((size_t)B_ * 4096 * 4);
  float*  g2   = (float*)alloc((size_t)B_ * NHID * 4);
  (void)ws_size; (void)in_sizes; (void)n_in; (void)out_size;

  k_cvt<<<NHID * NHID / 1024, 256, 0, stream>>>(WKw, WKbf);
  k_cvt<<<NHID * NHID / 1024, 256, 0, stream>>>(WVw, WVbf);
  k_qproj<<<128, 256, 0, stream>>>(x, WQw, WQb, q);

  dim3 gp(L_ / 128, NHID / 128, B_);                     // (16, 8, 32)
  k_proj<<<gp, 256, 0, stream>>>(WKbf, WKb, y, Kb);
  k_proj<<<gp, 256, 0, stream>>>(WVbf, WVb, y, Vb);

  k_attn<<<dim3(NHEAD, B_), 256, 0, stream>>>(q, Kb, Vb, x, z);
  k_bn<<<4, 256, 0, stream>>>(z, nullptr, nullptr, bn1g, bn1b, ret);

  k_mlp<1024, 4096, true ><<<64, 256, 0, stream>>>(ret, w1w, w1b, h);
  k_mlp<4096, 1024, false><<<16, 256, 0, stream>>>(h, w2w, nullptr, g2);

  k_bn<<<4, 256, 0, stream>>>(g2, ret, w2b, bn2g, bn2b, (float*)d_out);
}
